// Protonet_65833258713474
// MI455X (gfx1250) — matrix operand order repaired and checked by reading, not compiled, not run on hardware
//
#include <hip/hip_runtime.h>

typedef __attribute__((ext_vector_type(16))) __bf16 v16bf;
typedef __attribute__((ext_vector_type(8)))  float  v8f;

constexpr int NCLS = 64, NSUP = 5, NQRY = 64;
constexpr int NS  = NCLS * NSUP;   // 320 support rows
constexpr int NQ  = NCLS * NQRY;   // 4096 query rows
constexpr int NR  = NS + NQ;       // 4416 total rows
constexpr int DIN = 1024;
constexpr int CC  = 128, WIN2 = 25;
constexpr int DD  = CC * WIN2;     // 3200

__device__ __forceinline__ unsigned short f32_bf16(float f) {
  unsigned int x = __float_as_uint(f);
  x += 0x7fffu + ((x >> 16) & 1u);     // round-to-nearest-even
  return (unsigned short)(x >> 16);
}

// ---------------------------------------------------------------- converts
__global__ __launch_bounds__(256) void k_convert_x(const float* __restrict__ xs,
                                                   const float* __restrict__ xq,
                                                   unsigned short* __restrict__ Xb) {
  int idx = blockIdx.x * 256 + threadIdx.x;
  if (idx >= NR * DIN) return;
  int row = idx >> 10, col = idx & 1023;
  float v = (row < NS) ? xs[row * DIN + col] : xq[(row - NS) * DIN + col];
  Xb[idx] = f32_bf16(v);
}

// W (KxN, f32) -> Bt (NxK, bf16) tiled transpose through LDS.
__global__ __launch_bounds__(256)
void k_convert_wT(const float* __restrict__ W, unsigned short* __restrict__ Bt,
                  int K, int N) {
  __shared__ float tile[32][33];
  const int k0 = blockIdx.x * 32, n0 = blockIdx.y * 32;
  const int tid = threadIdx.x;
  {
    int r = tid >> 3, c = (tid & 7) * 4;                 // coalesced along N
    float4 v = *(const float4*)(W + (size_t)(k0 + r) * N + n0 + c);
    tile[r][c + 0] = v.x; tile[r][c + 1] = v.y;
    tile[r][c + 2] = v.z; tile[r][c + 3] = v.w;
  }
  __syncthreads();
  {
    int n = tid >> 3, k = (tid & 7) * 4;                 // coalesced along K
    unsigned int lo = (unsigned int)f32_bf16(tile[k + 0][n]) |
                      ((unsigned int)f32_bf16(tile[k + 1][n]) << 16);
    unsigned int hi = (unsigned int)f32_bf16(tile[k + 2][n]) |
                      ((unsigned int)f32_bf16(tile[k + 3][n]) << 16);
    uint2 o; o.x = lo; o.y = hi;
    *(uint2*)(Bt + (size_t)(n0 + n) * K + k0 + k) = o;
  }
}

// ---------------------------------------------------------------- WMMA GEMM
// C(MxN) = A(MxK,bf16) @ Bt(NxK,bf16)^T + bias, optional ReLU.
// Block tile 64x128, 8 waves (2x4); each wave: 32x32 patch = 4 WMMAs/chunk.
// LDS row stride 40 ushorts: 16B-aligned b128 frag loads, conflict-free banks.
union BfFrag { v16bf v; uint4 q[2]; };

template<bool RELU, bool WRITE_BF16>
__global__ __launch_bounds__(256)
void wmma_gemm(const unsigned short* __restrict__ A,    // M x K
               const unsigned short* __restrict__ Bt,   // N x K (pre-transposed)
               const float* __restrict__ bias,
               float* __restrict__ Cf, int f32Rows,
               unsigned short* __restrict__ Cb,
               int M, int N, int K) {
  __shared__ __attribute__((aligned(16))) unsigned short As[64][40];
  __shared__ __attribute__((aligned(16))) unsigned short Bs[128][40];

  const int tid  = threadIdx.x;
  const int wave = tid >> 5, lane = tid & 31;
  const int hf = lane >> 4, l = lane & 15;
  const int wm = wave >> 2, wn = wave & 3;
  const int rowBase = blockIdx.x * 64;
  const int colBase = blockIdx.y * 128;

  // cooperative loaders: all 16B vector ops
  const int ar = tid >> 2, ac = (tid & 3) * 8;          // A: 64 rows x 8 elems
  const int bn = tid >> 1, bk = (tid & 1) * 16;         // B: 128 rows x 16 elems
  const unsigned short* aPtr = A  + (size_t)(rowBase + ar) * K + ac;
  const unsigned short* bPtr = Bt + (size_t)(colBase + bn) * K + bk;

  uint4 aReg  = *(const uint4*)(aPtr);
  uint4 bReg0 = *(const uint4*)(bPtr);
  uint4 bReg1 = *(const uint4*)(bPtr + 8);

  v8f acc00 = {}, acc01 = {}, acc10 = {}, acc11 = {};

  const int m0 = wm * 32 + l,      m1 = m0 + 16;
  const int n0 = wn * 32 + l,      n1 = n0 + 16;

  for (int k0 = 0; k0 < K; k0 += 32) {
    *(uint4*)&As[ar][ac]      = aReg;
    *(uint4*)&Bs[bn][bk]      = bReg0;
    *(uint4*)&Bs[bn][bk + 8]  = bReg1;
    __syncthreads();

    if (k0 + 32 < K) {                                   // prefetch next chunk
      aReg  = *(const uint4*)(aPtr + k0 + 32);
      bReg0 = *(const uint4*)(bPtr + k0 + 32);
      bReg1 = *(const uint4*)(bPtr + k0 + 40);
    }

    BfFrag a0, a1, b0, b1;
    // A frag: two contiguous 8-elem K runs per lane (ISA 16-bit A layout)
    a0.q[0] = *(const uint4*)&As[m0][8 * hf];
    a0.q[1] = *(const uint4*)&As[m0][16 + 8 * hf];
    a1.q[0] = *(const uint4*)&As[m1][8 * hf];
    a1.q[1] = *(const uint4*)&As[m1][16 + 8 * hf];
    // B frag: two contiguous 8-elem K runs per lane (same K order as A)
    b0.q[0] = *(const uint4*)&Bs[n0][8 * hf];
    b0.q[1] = *(const uint4*)&Bs[n0][16 + 8 * hf];
    b1.q[0] = *(const uint4*)&Bs[n1][8 * hf];
    b1.q[1] = *(const uint4*)&Bs[n1][16 + 8 * hf];

    acc00 = __builtin_amdgcn_wmma_f32_16x16x32_bf16(false, a0.v, false, b0.v,
                                                    (short)0, acc00, false, false);
    acc01 = __builtin_amdgcn_wmma_f32_16x16x32_bf16(false, a0.v, false, b1.v,
                                                    (short)0, acc01, false, false);
    acc10 = __builtin_amdgcn_wmma_f32_16x16x32_bf16(false, a1.v, false, b0.v,
                                                    (short)0, acc10, false, false);
    acc11 = __builtin_amdgcn_wmma_f32_16x16x32_bf16(false, a1.v, false, b1.v,
                                                    (short)0, acc11, false, false);
    __syncthreads();
  }

  // epilogue: C/D layout -> VGPR r holds M = r + 8*hf, lane l holds N col
#pragma unroll
  for (int tm = 0; tm < 2; ++tm) {
#pragma unroll
    for (int tn = 0; tn < 2; ++tn) {
      const v8f& acc = tm == 0 ? (tn == 0 ? acc00 : acc01)
                               : (tn == 0 ? acc10 : acc11);
      const int n = colBase + wn * 32 + tn * 16 + l;
      const float bv = bias[n];
#pragma unroll
      for (int r = 0; r < 8; ++r) {
        int row = rowBase + wm * 32 + tm * 16 + r + 8 * hf;
        float v = acc[r] + bv;
        if (RELU) v = v > 0.f ? v : 0.f;
        if (row < f32Rows) Cf[(size_t)row * N + n] = v;
        if (WRITE_BF16)    Cb[(size_t)row * N + n] = f32_bf16(v);
      }
    }
  }
}

// ---------------------------------------------------------------- attention branch
__global__ __launch_bounds__(256) void k_pool(const float* __restrict__ Zs,
                                              float* __restrict__ pooled) {
  __shared__ float red[256];
  int c = blockIdx.x, tid = threadIdx.x;
  float s = 0.f;
  for (int i = tid; i < NS * WIN2; i += 256) {
    int r = i / WIN2, j = i % WIN2;
    s += Zs[(size_t)r * DD + c * WIN2 + j];
  }
  red[tid] = s; __syncthreads();
  for (int st = 128; st > 0; st >>= 1) {
    if (tid < st) red[tid] += red[tid + st];
    __syncthreads();
  }
  if (tid == 0) pooled[c] = red[0] / (float)(NS * WIN2);
}

__global__ __launch_bounds__(128) void k_att(const float* __restrict__ pooled,
                                             const float* __restrict__ W_att,
                                             const float* __restrict__ b_att,
                                             float* __restrict__ wvec) {
  __shared__ float sm[CC];
  int c = threadIdx.x;
  float acc = b_att[c];
  for (int i = 0; i < CC; ++i) acc += pooled[i] * W_att[i * CC + c];
  sm[c] = acc; __syncthreads();
  float m = sm[0];
  for (int i = 1; i < CC; ++i) m = sm[i] > m ? sm[i] : m;
  float e = expf(acc - m);
  sm[c] = e; __syncthreads();
  float s = 0.f;
  for (int i = 0; i < CC; ++i) s += sm[i];
  float wc = e / s * (float)CC;
  for (int j = 0; j < WIN2; ++j) wvec[c * WIN2 + j] = wc;
}

// ---------------------------------------------------------------- prototypes / distances
__global__ __launch_bounds__(256) void k_proto(const float* __restrict__ Z,
                                               float* __restrict__ zproto) {
  int idx = blockIdx.x * 256 + threadIdx.x;
  if (idx >= NCLS * DD) return;
  int k = idx / DD, d = idx % DD;
  float s = 0.f;
  for (int ss = 0; ss < NSUP; ++ss) s += Z[(size_t)(k * NSUP + ss) * DD + d];
  zproto[idx] = s * (1.f / NSUP);
}

__global__ __launch_bounds__(256) void k_p2(const float* __restrict__ zproto,
                                            const float* __restrict__ wvec,
                                            float* __restrict__ p2) {
  __shared__ float red[256];
  int k = blockIdx.x, tid = threadIdx.x;
  float s = 0.f;
  for (int d = tid; d < DD; d += 256) {
    float v = zproto[(size_t)k * DD + d];
    s += v * v * wvec[d];
  }
  red[tid] = s; __syncthreads();
  for (int st = 128; st > 0; st >>= 1) {
    if (tid < st) red[tid] += red[tid + st];
    __syncthreads();
  }
  if (tid == 0) p2[k] = red[0];
}

__global__ __launch_bounds__(256) void k_query(const float* __restrict__ Z,
                                               const float* __restrict__ zproto,
                                               const float* __restrict__ wvec,
                                               const float* __restrict__ p2,
                                               float* __restrict__ lossrow,
                                               float* __restrict__ accrow) {
  __shared__ float zqw[DD];
  __shared__ float red[256];
  __shared__ float logits[NCLS];
  int q = blockIdx.x, tid = threadIdx.x;
  const float* zq = Z + (size_t)(NS + q) * DD;

  float part = 0.f;
  for (int d = tid; d < DD; d += 256) {
    float v = zq[d];
    float vw = v * wvec[d];
    zqw[d] = vw;
    part += v * vw;
  }
  red[tid] = part; __syncthreads();
  for (int st = 128; st > 0; st >>= 1) {
    if (tid < st) red[tid] += red[tid + st];
    __syncthreads();
  }
  float q2 = red[0];
  __syncthreads();

  int k = tid >> 2, j = tid & 3;
  const float* zp = zproto + (size_t)k * DD;
  float c = 0.f;
  for (int d = j; d < DD; d += 4) c += zqw[d] * zp[d];
  red[tid] = c; __syncthreads();
  if (j == 0) {
    float cr = red[tid] + red[tid + 1] + red[tid + 2] + red[tid + 3];
    float dist = q2 + p2[k] - 2.f * cr;
    logits[k] = -dist;
  }
  __syncthreads();

  if (tid == 0) {
    int target = q >> 6;
    float m = logits[0]; int am = 0;
    for (int i = 1; i < NCLS; ++i)
      if (logits[i] > m) { m = logits[i]; am = i; }
    float s = 0.f;
    for (int i = 0; i < NCLS; ++i) s += expf(logits[i] - m);
    float lse = m + logf(s);
    lossrow[q] = -(logits[target] - lse);
    accrow[q]  = (am == target) ? 1.f : 0.f;
  }
}

__global__ __launch_bounds__(256) void k_final(const float* __restrict__ lossrow,
                                               const float* __restrict__ accrow,
                                               float* __restrict__ out) {
  __shared__ float rl[256], ra[256];
  int tid = threadIdx.x;
  float sl = 0.f, sa = 0.f;
  for (int i = tid; i < NQ; i += 256) { sl += lossrow[i]; sa += accrow[i]; }
  rl[tid] = sl; ra[tid] = sa; __syncthreads();
  for (int st = 128; st > 0; st >>= 1) {
    if (tid < st) { rl[tid] += rl[tid + st]; ra[tid] += ra[tid + st]; }
    __syncthreads();
  }
  if (tid == 0) {
    out[0] = rl[0] / (float)NQ;
    out[1] = ra[0] / (float)NQ;
  }
}

// ---------------------------------------------------------------- launch
extern "C" void kernel_launch(void* const* d_in, const int* in_sizes, int n_in,
                              void* d_out, int out_size, void* d_ws, size_t ws_size,
                              hipStream_t stream) {
  (void)in_sizes; (void)n_in; (void)out_size; (void)ws_size;
  const float* xs      = (const float*)d_in[0];
  const float* xq      = (const float*)d_in[1];
  const float* W_share = (const float*)d_in[2];
  const float* b_share = (const float*)d_in[3];
  const float* W_att   = (const float*)d_in[4];
  const float* b_att   = (const float*)d_in[5];
  const float* W_fine  = (const float*)d_in[6];
  const float* b_fine  = (const float*)d_in[7];
  float* out = (float*)d_out;

  char* base = (char*)d_ws;
  size_t off = 0;
  auto alloc = [&](size_t bytes) -> char* {
    char* p = base + off;
    off = (off + bytes + 255) & ~(size_t)255;
    return p;
  };
  unsigned short* Xb     = (unsigned short*)alloc((size_t)NR * DIN * 2);   //  9.0 MB
  unsigned short* WSt    = (unsigned short*)alloc((size_t)DIN * DD * 2);   //  6.6 MB (DD x DIN)
  unsigned short* WFt    = (unsigned short*)alloc((size_t)DD * DD * 2);    // 20.5 MB (DD x DD)
  float*          Zs     = (float*)alloc((size_t)NS * DD * 4);             //  4.1 MB
  unsigned short* Zsb    = (unsigned short*)alloc((size_t)NR * DD * 2);    // 28.3 MB
  float*          Z      = (float*)alloc((size_t)NR * DD * 4);             // 56.5 MB
  float*          pooled = (float*)alloc(CC * 4);
  float*          wvec   = (float*)alloc(DD * 4);
  float*          zproto = (float*)alloc((size_t)NCLS * DD * 4);
  float*          p2     = (float*)alloc(NCLS * 4);
  float*          lossrw = (float*)alloc(NQ * 4);
  float*          accrw  = (float*)alloc(NQ * 4);

  // 1. bf16 conversions (weights transposed to N x K for vectorized B staging)
  k_convert_x<<<(NR * DIN) / 256, 256, 0, stream>>>(xs, xq, Xb);
  k_convert_wT<<<dim3(DIN / 32, DD / 32), 256, 0, stream>>>(W_share, WSt, DIN, DD);
  k_convert_wT<<<dim3(DD / 32, DD / 32), 256, 0, stream>>>(W_fine, WFt, DD, DD);

  // 2. z_share = relu(x @ W_share + b_share)
  wmma_gemm<true, true><<<dim3(NR / 64, DD / 128), 256, 0, stream>>>(
      Xb, WSt, b_share, Zs, NS, Zsb, NR, DD, DIN);

  // 3. attention branch -> feature weights w (3200)
  k_pool<<<CC, 256, 0, stream>>>(Zs, pooled);
  k_att<<<1, CC, 0, stream>>>(pooled, W_att, b_att, wvec);

  // 4. z = z_share @ W_fine + b_fine
  wmma_gemm<false, false><<<dim3(NR / 64, DD / 128), 256, 0, stream>>>(
      Zsb, WFt, b_fine, Z, NR, nullptr, NR, DD, DD);

  // 5. prototypes, p2, per-query distances + log-softmax, final reduce
  k_proto<<<(NCLS * DD) / 256, 256, 0, stream>>>(Z, zproto);
  k_p2<<<NCLS, 256, 0, stream>>>(zproto, wvec, p2);
  k_query<<<NQ, 256, 0, stream>>>(Z, zproto, wvec, p2, lossrw, accrw);
  k_final<<<1, 256, 0, stream>>>(lossrw, accrw, out);
}
